// DescriptorLoss_82214263980041
// MI455X (gfx1250) — matrix operand order repaired and checked by reading, not compiled, not run on hardware
//
#include <hip/hip_runtime.h>
#include <hip/hip_bf16.h>

typedef __attribute__((ext_vector_type(16))) _Float16 v16h;
typedef __attribute__((ext_vector_type(8)))  _Float16 v8h;
typedef __attribute__((ext_vector_type(8)))  float    v8f;

namespace dl {
constexpr int BATCH = 2;
constexpr int NPTS  = 5000;
constexpr int CH    = 256;
constexpr int HH    = 96;
constexpr int WW    = 96;
constexpr int NPAD  = 5024;           // 32 * 157 (32x32 macro-tiles)
constexpr int BT    = NPAD / 32;      // 157 macro-tiles per dim
constexpr float THR2 = 0.01f * 0.01f; // MATCH_THR^2
}

// ---- order-preserving float <-> uint encoding for atomicMax on floats ----
__device__ __forceinline__ unsigned encf(float f) {
    unsigned u = __float_as_uint(f);
    return (u & 0x80000000u) ? ~u : (u | 0x80000000u);
}
__device__ __forceinline__ float decf(unsigned u) {
    return (u & 0x80000000u) ? __uint_as_float(u & 0x7fffffffu)
                             : __uint_as_float(~u);
}

// exact, reproducible squared distance (same rounding in every kernel)
__device__ __forceinline__ float dist2(float ax, float ay, float bx, float by) {
    float dx = ax - bx, dy = ay - by;
    return __fmaf_rn(dx, dx, __fmul_rn(dy, dy));
}

// ---------------- init stats ----------------
__global__ void k_init(unsigned* rowmaxE, unsigned* colmaxE,
                       float* rowsum, float* colsum, float* accum) {
    int i = blockIdx.x * blockDim.x + threadIdx.x;
    if (i < dl::BATCH * dl::NPAD) {
        rowmaxE[i] = 0u;   // below enc(-inf)
        colmaxE[i] = 0u;
        rowsum[i]  = 0.0f;
        colsum[i]  = 0.0f;
    }
    if (i < 2) accum[i] = 0.0f;
}

// ---------------- bilinear descriptor sampling -> f16 ----------------
__global__ __launch_bounds__(256) void k_sample(const float* __restrict__ img,
                                                const float* __restrict__ kpts,
                                                _Float16* __restrict__ out) {
    using namespace dl;
    int n = blockIdx.x;           // 0..NPAD-1
    int b = blockIdx.y;
    int c = threadIdx.x;          // 0..255
    size_t o = ((size_t)b * NPAD + n) * CH + c;
    if (n >= NPTS) { out[o] = (_Float16)0.0f; return; }

    float px = kpts[((size_t)b * NPTS + n) * 2 + 0];
    float py = kpts[((size_t)b * NPTS + n) * 2 + 1];
    float x = ((px + 1.0f) * (float)WW - 1.0f) * 0.5f;
    float y = ((py + 1.0f) * (float)HH - 1.0f) * 0.5f;
    float x0f = floorf(x), y0f = floorf(y);
    float wx = x - x0f,  wy = y - y0f;
    int x0 = (int)x0f, y0 = (int)y0f;

    const float* base = img + ((size_t)b * CH + c) * (size_t)(HH * WW);
    auto g = [&](int ix, int iy) -> float {
        if (ix < 0 || ix >= WW || iy < 0 || iy >= HH) return 0.0f;
        return base[(size_t)iy * WW + ix];
    };
    float v = (1.0f - wy) * ((1.0f - wx) * g(x0, y0)     + wx * g(x0 + 1, y0))
            +         wy  * ((1.0f - wx) * g(x0, y0 + 1) + wx * g(x0 + 1, y0 + 1));
    out[o] = (_Float16)v;
}

// ---------------- row/col nearest-neighbor distances ----------------
// z==0: rowminD[n] = min_m dist2(kAB[n], kB[m])
// z==1: colminD[m] = min_n dist2(kBA[m], kA[n])
__global__ __launch_bounds__(256) void k_mindist(const float* __restrict__ kA,
                                                 const float* __restrict__ kB,
                                                 const float* __restrict__ kAB,
                                                 const float* __restrict__ kBA,
                                                 float* __restrict__ rowminD,
                                                 float* __restrict__ colminD) {
    using namespace dl;
    int n = blockIdx.x, b = blockIdx.y, role = blockIdx.z;
    const float* P = role ? kBA : kAB;
    const float* Q = role ? kA  : kB;
    float px = P[((size_t)b * NPTS + n) * 2 + 0];
    float py = P[((size_t)b * NPTS + n) * 2 + 1];
    float best = 3.0e38f;
    for (int m = threadIdx.x; m < NPTS; m += blockDim.x) {
        float qx = Q[((size_t)b * NPTS + m) * 2 + 0];
        float qy = Q[((size_t)b * NPTS + m) * 2 + 1];
        best = fminf(best, dist2(px, py, qx, qy));
    }
    __shared__ float sm[256];
    sm[threadIdx.x] = best;
    __syncthreads();
    for (int s = 128; s > 0; s >>= 1) {
        if (threadIdx.x < s) sm[threadIdx.x] = fminf(sm[threadIdx.x], sm[threadIdx.x + s]);
        __syncthreads();
    }
    if (threadIdx.x == 0)
        (role ? colminD : rowminD)[(size_t)b * NPTS + n] = sm[0];
}

// ---------------- cross-lane helpers (wave32) ----------------
__device__ __forceinline__ float redMax16(float v) {  // across each 16-lane half
    v = fmaxf(v, __shfl_xor(v, 1, 32));
    v = fmaxf(v, __shfl_xor(v, 2, 32));
    v = fmaxf(v, __shfl_xor(v, 4, 32));
    v = fmaxf(v, __shfl_xor(v, 8, 32));
    return v;
}
__device__ __forceinline__ float redSum16(float v) {
    v += __shfl_xor(v, 1, 32);
    v += __shfl_xor(v, 2, 32);
    v += __shfl_xor(v, 4, 32);
    v += __shfl_xor(v, 8, 32);
    return v;
}

// ---------------- fragment loaders ----------------
// A (16x32 f16): lane<16 holds K {0..7,16..23} of row l; lane>=16 K {8..15,24..31}
__device__ __forceinline__ v16h ldA(const _Float16* __restrict__ p, int kk, int half) {
    v8h x0 = *(const v8h*)(p + kk + half * 8);
    v8h x1 = *(const v8h*)(p + kk + 16 + half * 8);
    return __builtin_shufflevector(x0, x1, 0,1,2,3,4,5,6,7,8,9,10,11,12,13,14,15);
}
// B (32x16 f16): lane<16 holds K {0..7,16..23} of col l; lane>=16 K {8..15,24..31}
__device__ __forceinline__ v16h ldB(const _Float16* __restrict__ p, int kk, int half) {
    v8h x0 = *(const v8h*)(p + kk + half * 8);
    v8h x1 = *(const v8h*)(p + kk + 16 + half * 8);
    return __builtin_shufflevector(x0, x1, 0,1,2,3,4,5,6,7,8,9,10,11,12,13,14,15);
}

// ---------------- WMMA GEMM passes (32x32 region per wave, 2x2 tiles) ----------
// MODE 0: row/col max of corr      MODE 1: row/col sum of exp(corr - max)
// MODE 2: masked NLL accumulation
template <int MODE>
__global__ __launch_bounds__(256) void k_gemm(const _Float16* __restrict__ dA,
                                              const _Float16* __restrict__ dB,
                                              unsigned* __restrict__ rowmaxE,
                                              unsigned* __restrict__ colmaxE,
                                              float* __restrict__ rowsum,
                                              float* __restrict__ colsum,
                                              const float* __restrict__ rowminD,
                                              const float* __restrict__ colminD,
                                              const float* __restrict__ kA,
                                              const float* __restrict__ kB,
                                              const float* __restrict__ kAB,
                                              const float* __restrict__ kBA,
                                              float* __restrict__ accum) {
    using namespace dl;
    const int wave = threadIdx.x >> 5;
    const int lane = threadIdx.x & 31;
    const long tileId = (long)blockIdx.x * (blockDim.x >> 5) + wave;
    const int b = blockIdx.y;
    if (tileId >= (long)BT * BT) return;         // wave-uniform: EXEC stays all-ones
    const int nBase = (int)(tileId / BT) * 32;
    const int mBase = (int)(tileId % BT) * 32;
    const int half = lane >> 4, l = lane & 15;
    const size_t bo = (size_t)b * NPAD;

    const _Float16* aRow0 = dA + (bo + nBase + l) * CH;
    const _Float16* aRow1 = aRow0 + (size_t)16 * CH;
    const _Float16* bRow0 = dB + (bo + mBase + l) * CH;
    const _Float16* bRow1 = bRow0 + (size_t)16 * CH;

    v8f acc[2][2] = {};
#pragma unroll
    for (int kk = 0; kk < CH; kk += 32) {
        v16h a0 = ldA(aRow0, kk, half);
        v16h a1 = ldA(aRow1, kk, half);
        v16h b0 = ldB(bRow0, kk, half);
        v16h b1 = ldB(bRow1, kk, half);
        acc[0][0] = __builtin_amdgcn_wmma_f32_16x16x32_f16(false, a0, false, b0,
                                                           (short)0, acc[0][0], false, false);
        acc[0][1] = __builtin_amdgcn_wmma_f32_16x16x32_f16(false, a0, false, b1,
                                                           (short)0, acc[0][1], false, false);
        acc[1][0] = __builtin_amdgcn_wmma_f32_16x16x32_f16(false, a1, false, b0,
                                                           (short)0, acc[1][0], false, false);
        acc[1][1] = __builtin_amdgcn_wmma_f32_16x16x32_f16(false, a1, false, b1,
                                                           (short)0, acc[1][1], false, false);
    }

    // D layout per tile: lane L, element r -> M = r + (L>=16 ? 8 : 0), N = L&15
    bool cOK[2] = { (mBase + l) < NPTS, (mBase + 16 + l) < NPTS };

    if (MODE == 0) {
        float colAcc[2] = { -3.0e38f, -3.0e38f };
#pragma unroll
        for (int i = 0; i < 2; i++) {
#pragma unroll
            for (int r = 0; r < 8; r++) {
                int row = nBase + i * 16 + half * 8 + r;
                bool rOK = row < NPTS;
                float v0 = (rOK && cOK[0]) ? acc[i][0][r] : -3.0e38f;
                float v1 = (rOK && cOK[1]) ? acc[i][1][r] : -3.0e38f;
                float rv = redMax16(fmaxf(v0, v1));
                if (l == 0 && rOK) atomicMax(&rowmaxE[bo + row], encf(rv));
                colAcc[0] = fmaxf(colAcc[0], v0);
                colAcc[1] = fmaxf(colAcc[1], v1);
            }
        }
#pragma unroll
        for (int j = 0; j < 2; j++) {
            float cm = fmaxf(colAcc[j], __shfl_xor(colAcc[j], 16, 32));
            if (half == 0 && cOK[j])
                atomicMax(&colmaxE[bo + mBase + j * 16 + l], encf(cm));
        }
    } else if (MODE == 1) {
        float cmax[2] = { decf(colmaxE[bo + mBase + l]),           // idx < NPAD: safe
                          decf(colmaxE[bo + mBase + 16 + l]) };
        float colAcc[2] = { 0.0f, 0.0f };
#pragma unroll
        for (int i = 0; i < 2; i++) {
#pragma unroll
            for (int r = 0; r < 8; r++) {
                int row = nBase + i * 16 + half * 8 + r;
                bool rOK = row < NPTS;
                float rmax = decf(rowmaxE[bo + row]);
                float e0 = (rOK && cOK[0]) ? __expf(acc[i][0][r] - rmax) : 0.0f;
                float e1 = (rOK && cOK[1]) ? __expf(acc[i][1][r] - rmax) : 0.0f;
                float sr = redSum16(e0 + e1);
                if (l == 0 && rOK) atomicAdd(&rowsum[bo + row], sr);
                colAcc[0] += (rOK && cOK[0]) ? __expf(acc[i][0][r] - cmax[0]) : 0.0f;
                colAcc[1] += (rOK && cOK[1]) ? __expf(acc[i][1][r] - cmax[1]) : 0.0f;
            }
        }
#pragma unroll
        for (int j = 0; j < 2; j++) {
            float cs = colAcc[j] + __shfl_xor(colAcc[j], 16, 32);
            if (half == 0 && cOK[j])
                atomicAdd(&colsum[bo + mBase + j * 16 + l], cs);
        }
    } else {
        float lseC[2] = { 0.0f, 0.0f }, colD[2] = { 1.0e30f, 1.0e30f };
        float kBx[2] = {}, kBy[2] = {}, kBAx[2] = {}, kBAy[2] = {};
#pragma unroll
        for (int j = 0; j < 2; j++) {
            int col = mBase + j * 16 + l;
            if (cOK[j]) {
                lseC[j] = decf(colmaxE[bo + col]) + __logf(colsum[bo + col]);
                colD[j] = colminD[(size_t)b * NPTS + col];
                kBx[j]  = kB [((size_t)b * NPTS + col) * 2 + 0];
                kBy[j]  = kB [((size_t)b * NPTS + col) * 2 + 1];
                kBAx[j] = kBA[((size_t)b * NPTS + col) * 2 + 0];
                kBAy[j] = kBA[((size_t)b * NPTS + col) * 2 + 1];
            }
        }
        float s = 0.0f, cnt = 0.0f;
#pragma unroll
        for (int i = 0; i < 2; i++) {
#pragma unroll
            for (int r = 0; r < 8; r++) {
                int row = nBase + i * 16 + half * 8 + r;
                bool rOK = row < NPTS;
                if (rOK) {
                    float lseR = decf(rowmaxE[bo + row]) + __logf(rowsum[bo + row]);
                    float rmD  = rowminD[(size_t)b * NPTS + row];
                    float ax = kAB[((size_t)b * NPTS + row) * 2 + 0];
                    float ay = kAB[((size_t)b * NPTS + row) * 2 + 1];
                    float qx = kA [((size_t)b * NPTS + row) * 2 + 0];
                    float qy = kA [((size_t)b * NPTS + row) * 2 + 1];
#pragma unroll
                    for (int j = 0; j < 2; j++) {
                        if (cOK[j]) {
                            float d2B = dist2(ax, ay, kBx[j], kBy[j]);    // role-0 rounding
                            float d2A = dist2(kBAx[j], kBAy[j], qx, qy);  // role-1 rounding
                            bool m = (d2B == rmD) && (d2A == colD[j]) &&
                                     (d2B < dl::THR2) && (d2A < dl::THR2);
                            if (m) {
                                s += 2.0f * acc[i][j][r] - lseR - lseC[j];
                                cnt += 1.0f;
                            }
                        }
                    }
                }
            }
        }
        s   = redSum16(s);   s   += __shfl_xor(s, 16, 32);
        cnt = redSum16(cnt); cnt += __shfl_xor(cnt, 16, 32);
        if (lane == 0) {
            atomicAdd(&accum[0], s);
            atomicAdd(&accum[1], cnt);
        }
    }
}

__global__ void k_fin(const float* __restrict__ accum, float* __restrict__ out) {
    out[0] = -accum[0] / fmaxf(accum[1], 1.0f);
}

extern "C" void kernel_launch(void* const* d_in, const int* in_sizes, int n_in,
                              void* d_out, int out_size, void* d_ws, size_t ws_size,
                              hipStream_t stream) {
    using namespace dl;
    (void)in_sizes; (void)n_in; (void)out_size; (void)ws_size;

    const float* gridA = (const float*)d_in[0];
    const float* gridB = (const float*)d_in[1];
    const float* kA    = (const float*)d_in[2];
    const float* kB    = (const float*)d_in[3];
    const float* kAB   = (const float*)d_in[4];
    const float* kBA   = (const float*)d_in[5];
    float* out = (float*)d_out;

    size_t off = 0;
    auto take = [&](size_t bytes) -> char* {
        char* p = (char*)d_ws + off;
        off = (off + bytes + 255) & ~(size_t)255;
        return p;
    };
    _Float16* descA  = (_Float16*)take((size_t)BATCH * NPAD * CH * sizeof(_Float16));
    _Float16* descB  = (_Float16*)take((size_t)BATCH * NPAD * CH * sizeof(_Float16));
    unsigned* rowmaxE = (unsigned*)take((size_t)BATCH * NPAD * sizeof(unsigned));
    unsigned* colmaxE = (unsigned*)take((size_t)BATCH * NPAD * sizeof(unsigned));
    float* rowsum  = (float*)take((size_t)BATCH * NPAD * sizeof(float));
    float* colsum  = (float*)take((size_t)BATCH * NPAD * sizeof(float));
    float* rowminD = (float*)take((size_t)BATCH * NPTS * sizeof(float));
    float* colminD = (float*)take((size_t)BATCH * NPTS * sizeof(float));
    float* accum   = (float*)take(256);

    k_init<<<dim3((BATCH * NPAD + 255) / 256), 256, 0, stream>>>(
        rowmaxE, colmaxE, rowsum, colsum, accum);

    k_sample<<<dim3(NPAD, BATCH), 256, 0, stream>>>(gridA, kA, descA);
    k_sample<<<dim3(NPAD, BATCH), 256, 0, stream>>>(gridB, kB, descB);

    k_mindist<<<dim3(NPTS, BATCH, 2), 256, 0, stream>>>(
        kA, kB, kAB, kBA, rowminD, colminD);

    dim3 ggrid((unsigned)(((long)BT * BT + 7) / 8), BATCH);
    k_gemm<0><<<ggrid, 256, 0, stream>>>(descA, descB, rowmaxE, colmaxE,
                                         rowsum, colsum, rowminD, colminD,
                                         kA, kB, kAB, kBA, accum);
    k_gemm<1><<<ggrid, 256, 0, stream>>>(descA, descB, rowmaxE, colmaxE,
                                         rowsum, colsum, rowminD, colminD,
                                         kA, kB, kAB, kBA, accum);
    k_gemm<2><<<ggrid, 256, 0, stream>>>(descA, descB, rowmaxE, colmaxE,
                                         rowsum, colsum, rowminD, colminD,
                                         kA, kB, kAB, kBA, accum);

    k_fin<<<1, 1, 0, stream>>>(accum, out);
}
